// EncodingLayer_58506044506747
// MI455X (gfx1250) — matrix operand order repaired and checked by reading, not compiled, not run on hardware
//
#include <hip/hip_runtime.h>
#include <hip/hip_bf16.h>

typedef _Float16 f16;
typedef __attribute__((ext_vector_type(16))) _Float16 v16h;
typedef __attribute__((ext_vector_type(8)))  _Float16 v8h;
typedef __attribute__((ext_vector_type(8)))  float    v8f;

#define DEV static __device__ __forceinline__

static constexpr int B_ = 4, S_ = 2048, D_ = 512, H_ = 8, K_ = 64, FF_ = 64;
static constexpr int M_ = B_ * S_;   // 8192 token rows

// ---------------------------------------------------------------- WMMA core
DEV v8f wmma16(v16h a, v16h b, v8f c) {
  return __builtin_amdgcn_wmma_f32_16x16x32_f16(false, a, false, b, (short)0, c, false, false);
}

// A fragment 16x32 (MxK) f16, row-major source, ld along K.
// element e: K = (e&7) + 8*half + 16*(e>>3); row M = lane&15 (half = lane>>4)
DEV v16h fragA_f16(const f16* __restrict__ p, int ld, int m, int k0, int lane) {
  const int half = lane >> 4;
  const f16* r = p + (size_t)(m + (lane & 15)) * ld + k0 + 8 * half;
  v8h lo = *(const v8h*)r;
  v8h hi = *(const v8h*)(r + 16);
  v16h o;
#pragma unroll
  for (int i = 0; i < 8; ++i) { o[i] = lo[i]; o[8 + i] = hi[i]; }
  return o;
}

// Same fragment converting from fp32 source.
DEV v16h fragA_f32(const float* __restrict__ p, int ld, int m, int k0, int lane) {
  const int half = lane >> 4;
  const float* r = p + (size_t)(m + (lane & 15)) * ld + k0 + 8 * half;
  v16h o;
#pragma unroll
  for (int i = 0; i < 8; ++i) { o[i] = (f16)r[i]; o[8 + i] = (f16)r[16 + i]; }
  return o;
}

// A fragment from an LDS-staged fp32 tile; aw already points at row*36 + 8*half.
DEV v16h fragA_lds(const float* aw) {
  v16h o;
#pragma unroll
  for (int i = 0; i < 8; ++i) { o[i] = (f16)aw[i]; o[8 + i] = (f16)aw[16 + i]; }
  return o;
}

// B fragment 32x16 (KxN) f16 from transposed store Bt[n][k].
// element e: K = (e&7) + 8*half + 16*(e>>3); col N = lane&15 (half = lane>>4)
DEV v16h fragB_f16(const f16* __restrict__ p, int ld, int n, int k0, int lane) {
  const f16* r = p + (size_t)(n + (lane & 15)) * ld + k0 + 8 * (lane >> 4);
  v8h lo = *(const v8h*)r;
  v8h hi = *(const v8h*)(r + 16);
  v16h o;
#pragma unroll
  for (int i = 0; i < 8; ++i) { o[i] = lo[i]; o[8 + i] = hi[i]; }
  return o;
}

// --------------------------------------------------- weight cvt + transpose
__global__ void k_wcvt(const float* __restrict__ W, f16* __restrict__ Wt, int Kd, int N) {
  int i = blockIdx.x * 256 + threadIdx.x;
  if (i < Kd * N) {
    int n = i / Kd, k = i - n * Kd;
    Wt[i] = (f16)W[(size_t)k * N + n];
  }
}

// --------------------------------------------------------- QKV projections
// Each wave owns a unique 16-row M-subtile and sweeps 4 N-tiles (A reuse 4x).
// VSTORE=0: row-major f16 [m][n]; VSTORE=1: vT[b][h][kk][s] (for ctx B frags).
template <int VSTORE>
__global__ void k_proj(const float* __restrict__ X, const f16* __restrict__ Wt,
                       const float* __restrict__ bias, f16* __restrict__ out) {
  const int lane = threadIdx.x & 31, wave = threadIdx.x >> 5;
  const int m0 = blockIdx.x * 128 + wave * 16;
  const int n0 = blockIdx.y * 64;
  v8f acc[4] = {};
#pragma unroll 2
  for (int k0 = 0; k0 < D_; k0 += 32) {
    v16h a = fragA_f32(X, D_, m0, k0, lane);
#pragma unroll
    for (int t = 0; t < 4; ++t) {
      v16h b = fragB_f16(Wt, D_, n0 + t * 16, k0, lane);
      acc[t] = wmma16(a, b, acc[t]);
    }
  }
  const int half = lane >> 4, col = lane & 15;
#pragma unroll
  for (int t = 0; t < 4; ++t) {
    int n = n0 + t * 16 + col;
    float bv = bias[n];
#pragma unroll
    for (int r = 0; r < 8; ++r) {
      int m = m0 + r + 8 * half;
      float v = acc[t][r] + bv;
      if (VSTORE == 0) {
        out[(size_t)m * (H_ * K_) + n] = (f16)v;
      } else {                       // V stored transposed: vT[b][h][kk][s]
        int b = m >> 11, s = m & (S_ - 1);
        int h = n >> 6, kk = n & 63;
        out[(((size_t)b * H_ + h) * K_ + kk) * S_ + s] = (f16)v;
      }
    }
  }
}

// ------------------------------------------------------------ scores = QK^T
__global__ void k_scores(const f16* __restrict__ Q, const f16* __restrict__ Kt,
                         float* __restrict__ Atn) {
  const int lane = threadIdx.x & 31, wave = threadIdx.x >> 5;
  const int bh = blockIdx.z, b = bh >> 3, h = bh & 7;
  const int m0 = blockIdx.y * 128 + wave * 16;
  const int n0 = blockIdx.x * 64;
  const f16* q  = Q  + (size_t)b * S_ * (H_ * K_) + h * K_;
  const f16* kp = Kt + (size_t)b * S_ * (H_ * K_) + h * K_;
  v8f acc[4] = {};
#pragma unroll
  for (int k0 = 0; k0 < K_; k0 += 32) {
    v16h a = fragA_f16(q, H_ * K_, m0, k0, lane);
#pragma unroll
    for (int t = 0; t < 4; ++t) {
      v16h bfrag = fragB_f16(kp, H_ * K_, n0 + t * 16, k0, lane);
      acc[t] = wmma16(a, bfrag, acc[t]);
    }
  }
  float* out = Atn + (size_t)(h * B_ + b) * S_ * S_;
  const int half = lane >> 4, col = lane & 15;
#pragma unroll
  for (int t = 0; t < 4; ++t)
#pragma unroll
    for (int r = 0; r < 8; ++r)
      out[(size_t)(m0 + r + 8 * half) * S_ + n0 + t * 16 + col] = acc[t][r] * 0.125f;
}

// ------------------------------------------------------------- row softmax
__global__ void k_softmax(float* __restrict__ Atn) {
  __shared__ float sm[8];
  float* row = Atn + (size_t)blockIdx.x * S_;
  const int t = threadIdx.x, lane = t & 31, wave = t >> 5;
  float v[8];
  float mx = -3.4e38f;
#pragma unroll
  for (int i = 0; i < 8; ++i) { v[i] = row[t + 256 * i]; mx = fmaxf(mx, v[i]); }
#pragma unroll
  for (int o = 16; o > 0; o >>= 1) mx = fmaxf(mx, __shfl_xor(mx, o, 32));
  if (lane == 0) sm[wave] = mx;
  __syncthreads();
  mx = fmaxf(fmaxf(fmaxf(sm[0], sm[1]), fmaxf(sm[2], sm[3])),
             fmaxf(fmaxf(sm[4], sm[5]), fmaxf(sm[6], sm[7])));
  __syncthreads();
  float s = 0.f;
#pragma unroll
  for (int i = 0; i < 8; ++i) { v[i] = __expf(v[i] - mx); s += v[i]; }
#pragma unroll
  for (int o = 16; o > 0; o >>= 1) s += __shfl_xor(s, o, 32);
  if (lane == 0) sm[wave] = s;
  __syncthreads();
  s = sm[0] + sm[1] + sm[2] + sm[3] + sm[4] + sm[5] + sm[6] + sm[7];
  const float inv = 1.f / s;
#pragma unroll
  for (int i = 0; i < 8; ++i) row[t + 256 * i] = v[i] * inv;
}

// ------------------------------------------------------------ ctx = atn @ V
// The atn stream (537 MB fp32) is double-buffered into LDS with CDNA5 async
// global->LDS copies, synchronized with s_wait_asynccnt (per-wave, in-order).
DEV void async_tileA(const float* __restrict__ gbase, unsigned ldsbyte, int lane) {
#pragma unroll
  for (int c = 0; c < 4; ++c) {
    int id = lane + 32 * c;               // 128 chunks of 16B = 16 rows x 32 fp32
    int row = id >> 3, col4 = id & 7;
    unsigned long long g = (unsigned long long)(const void*)(gbase + (size_t)row * S_ + col4 * 4);
    unsigned l = ldsbyte + (unsigned)(row * 144 + col4 * 16);
    asm volatile("global_load_async_to_lds_b128 %0, %1, off"
                 :: "v"(l), "v"(g) : "memory");
  }
}

__global__ void k_ctx(const float* __restrict__ Atn, const f16* __restrict__ Vt,
                      f16* __restrict__ Ctx) {
  __shared__ float lds_a[2][8][16 * 36];     // 2 bufs x 8 waves x padded 16x32 tile
  const int lane = threadIdx.x & 31, wave = threadIdx.x >> 5;
  const int bh = blockIdx.z, b = bh >> 3, h = bh & 7;
  const float* A = Atn + (size_t)(h * B_ + b) * S_ * S_;
  const f16*  vt = Vt + (size_t)(b * H_ + h) * K_ * S_;   // [kk][s]
  const int m0 = blockIdx.x * 128 + wave * 16;
  const unsigned ldsb0 = (unsigned)((0 * 8 + wave) * 2304);
  const unsigned ldsb1 = (unsigned)((1 * 8 + wave) * 2304);

  v8f acc[4] = {};
  async_tileA(A + (size_t)m0 * S_, ldsb0, lane);
  int pb = 0;
  for (int k0 = 0; k0 < S_; k0 += 32) {
    if (k0 + 32 < S_) {
      async_tileA(A + (size_t)m0 * S_ + k0 + 32, pb ? ldsb0 : ldsb1, lane);
      asm volatile("s_wait_asynccnt 0x4" ::: "memory");   // current tile landed
    } else {
      asm volatile("s_wait_asynccnt 0x0" ::: "memory");
    }
    const float* aw = &lds_a[pb][wave][(lane & 15) * 36 + 8 * (lane >> 4)];
    v16h a = fragA_lds(aw);
#pragma unroll
    for (int t = 0; t < 4; ++t) {
      v16h bfrag = fragB_f16(vt, S_, t * 16, k0, lane);
      acc[t] = wmma16(a, bfrag, acc[t]);
    }
    pb ^= 1;
  }
  const int half = lane >> 4, col = lane & 15;
#pragma unroll
  for (int t = 0; t < 4; ++t)
#pragma unroll
    for (int r = 0; r < 8; ++r)
      Ctx[(size_t)(m0 + r + 8 * half) * (H_ * K_) + h * K_ + t * 16 + col] = (f16)acc[t][r];
}

// ------------------------------------- o = ctx @ Wo + bo + x  (fp32 out)
__global__ void k_oproj(const f16* __restrict__ Ctx, const f16* __restrict__ WoT,
                        const float* __restrict__ bo, const float* __restrict__ X,
                        float* __restrict__ Oplus) {
  const int lane = threadIdx.x & 31, wave = threadIdx.x >> 5;
  const int m0 = blockIdx.x * 128 + wave * 16;
  const int n0 = blockIdx.y * 64;
  v8f acc[4] = {};
#pragma unroll 2
  for (int k0 = 0; k0 < D_; k0 += 32) {
    v16h a = fragA_f16(Ctx, D_, m0, k0, lane);
#pragma unroll
    for (int t = 0; t < 4; ++t) {
      v16h b = fragB_f16(WoT, D_, n0 + t * 16, k0, lane);
      acc[t] = wmma16(a, b, acc[t]);
    }
  }
  const int half = lane >> 4, col = lane & 15;
#pragma unroll
  for (int t = 0; t < 4; ++t) {
    int n = n0 + t * 16 + col;
    float bv = bo[n];
#pragma unroll
    for (int r = 0; r < 8; ++r) {
      size_t m = m0 + r + 8 * half;
      Oplus[m * D_ + n] = acc[t][r] + bv + X[m * D_ + n];
    }
  }
}

// ----------------------------------------------- block-wide sum over 256 thr
DEV float bsum(float v, float* sm) {
#pragma unroll
  for (int o = 16; o > 0; o >>= 1) v += __shfl_xor(v, o, 32);
  const int lane = threadIdx.x & 31, wave = threadIdx.x >> 5;
  if (lane == 0) sm[wave] = v;
  __syncthreads();
  float t = sm[0] + sm[1] + sm[2] + sm[3] + sm[4] + sm[5] + sm[6] + sm[7];
  __syncthreads();
  return t;
}

// mha = LN(Oplus); h = LN(mha + x); store h fp32 + f16
__global__ void k_ln2(const float* __restrict__ Oplus, const float* __restrict__ X,
                      const float* __restrict__ gm, const float* __restrict__ bm,
                      const float* __restrict__ g1, const float* __restrict__ b1,
                      float* __restrict__ Hf, f16* __restrict__ Hc) {
  __shared__ float sm[8];
  const size_t row = blockIdx.x;
  const float* o = Oplus + row * D_;
  const float* x = X + row * D_;
  const int t = threadIdx.x;
  float a0 = o[t], a1 = o[t + 256];
  float mean = bsum(a0 + a1, sm) * (1.f / D_);
  float d0 = a0 - mean, d1 = a1 - mean;
  float var = bsum(d0 * d0 + d1 * d1, sm) * (1.f / D_);
  float ri = rsqrtf(var + 1e-5f);
  float h0 = d0 * ri * gm[t]       + bm[t]       + x[t];
  float h1 = d1 * ri * gm[t + 256] + bm[t + 256] + x[t + 256];
  mean = bsum(h0 + h1, sm) * (1.f / D_);
  d0 = h0 - mean; d1 = h1 - mean;
  var = bsum(d0 * d0 + d1 * d1, sm) * (1.f / D_);
  ri = rsqrtf(var + 1e-5f);
  float r0 = d0 * ri * g1[t]       + b1[t];
  float r1 = d1 * ri * g1[t + 256] + b1[t + 256];
  Hf[row * D_ + t] = r0;        Hf[row * D_ + t + 256] = r1;
  Hc[row * D_ + t] = (f16)r0;   Hc[row * D_ + t + 256] = (f16)r1;
}

// ------------------------------------------- t = relu(h @ W1 + b1)  (f16)
__global__ void k_ffn1(const f16* __restrict__ Hc, const f16* __restrict__ W1T,
                       const float* __restrict__ b1, f16* __restrict__ T) {
  const int lane = threadIdx.x & 31, wave = threadIdx.x >> 5;
  const int m0 = blockIdx.x * 128 + wave * 16;
  v8f acc[4] = {};
#pragma unroll 2
  for (int k0 = 0; k0 < D_; k0 += 32) {
    v16h a = fragA_f16(Hc, D_, m0, k0, lane);
#pragma unroll
    for (int t = 0; t < 4; ++t) {
      v16h b = fragB_f16(W1T, D_, t * 16, k0, lane);
      acc[t] = wmma16(a, b, acc[t]);
    }
  }
  const int half = lane >> 4, col = lane & 15;
#pragma unroll
  for (int t = 0; t < 4; ++t) {
    int n = t * 16 + col;
    float bv = b1[n];
#pragma unroll
    for (int r = 0; r < 8; ++r)
      T[(size_t)(m0 + r + 8 * half) * FF_ + n] = (f16)fmaxf(acc[t][r] + bv, 0.f);
  }
}

// ---------------------------------- ffres = t @ W2 + b2 + h  (fp32)
__global__ void k_ffn2(const f16* __restrict__ T, const f16* __restrict__ W2T,
                       const float* __restrict__ b2, const float* __restrict__ Hf,
                       float* __restrict__ R) {
  const int lane = threadIdx.x & 31, wave = threadIdx.x >> 5;
  const int m0 = blockIdx.x * 128 + wave * 16;
  const int n0 = blockIdx.y * 64;
  v8f acc[4] = {};
#pragma unroll
  for (int k0 = 0; k0 < FF_; k0 += 32) {
    v16h a = fragA_f16(T, FF_, m0, k0, lane);
#pragma unroll
    for (int t = 0; t < 4; ++t) {
      v16h b = fragB_f16(W2T, FF_, n0 + t * 16, k0, lane);
      acc[t] = wmma16(a, b, acc[t]);
    }
  }
  const int half = lane >> 4, col = lane & 15;
#pragma unroll
  for (int t = 0; t < 4; ++t) {
    int n = n0 + t * 16 + col;
    float bv = b2[n];
#pragma unroll
    for (int r = 0; r < 8; ++r) {
      size_t m = m0 + r + 8 * half;
      R[m * D_ + n] = acc[t][r] + bv + Hf[m * D_ + n];
    }
  }
}

// out = LN(R)  -> d_out
__global__ void k_lnf(const float* __restrict__ R, const float* __restrict__ g,
                      const float* __restrict__ b, float* __restrict__ Out) {
  __shared__ float sm[8];
  const size_t row = blockIdx.x;
  const float* r = R + row * D_;
  const int t = threadIdx.x;
  float a0 = r[t], a1 = r[t + 256];
  float mean = bsum(a0 + a1, sm) * (1.f / D_);
  float d0 = a0 - mean, d1 = a1 - mean;
  float var = bsum(d0 * d0 + d1 * d1, sm) * (1.f / D_);
  float ri = rsqrtf(var + 1e-5f);
  Out[row * D_ + t]       = d0 * ri * g[t]       + b[t];
  Out[row * D_ + t + 256] = d1 * ri * g[t + 256] + b[t + 256];
}

// ---------------------------------------------------------------- launcher
extern "C" void kernel_launch(void* const* d_in, const int* in_sizes, int n_in,
                              void* d_out, int out_size, void* d_ws, size_t ws_size,
                              hipStream_t stream) {
  const float* x   = (const float*)d_in[0];
  const float* Wq  = (const float*)d_in[1];
  const float* bq  = (const float*)d_in[2];
  const float* Wk  = (const float*)d_in[3];
  const float* bk  = (const float*)d_in[4];
  const float* Wv  = (const float*)d_in[5];
  const float* bv  = (const float*)d_in[6];
  const float* Wo  = (const float*)d_in[7];
  const float* bo  = (const float*)d_in[8];
  const float* gmh = (const float*)d_in[9];
  const float* bmh = (const float*)d_in[10];
  const float* gn1 = (const float*)d_in[11];
  const float* bn1 = (const float*)d_in[12];
  const float* W1  = (const float*)d_in[13];
  const float* b1  = (const float*)d_in[14];
  const float* W2  = (const float*)d_in[15];
  const float* b2  = (const float*)d_in[16];
  const float* gn2 = (const float*)d_in[17];
  const float* bn2 = (const float*)d_in[18];

  float* out = (float*)d_out;                       // [M_, D_]
  float* atn = out + (size_t)M_ * D_;               // [H*B, S, S]

  char* w = (char*)d_ws;
  size_t off = 0;
  auto take = [&](size_t bytes) -> char* {
    char* p = w + off;
    off += (bytes + 255) & ~(size_t)255;
    return p;
  };
  f16* wqT = (f16*)take((size_t)D_ * D_ * 2);
  f16* wkT = (f16*)take((size_t)D_ * D_ * 2);
  f16* wvT = (f16*)take((size_t)D_ * D_ * 2);
  f16* woT = (f16*)take((size_t)D_ * D_ * 2);
  f16* w1T = (f16*)take((size_t)FF_ * D_ * 2);
  f16* w2T = (f16*)take((size_t)D_ * FF_ * 2);
  f16* q16 = (f16*)take((size_t)M_ * D_ * 2);
  f16* k16 = (f16*)take((size_t)M_ * D_ * 2);
  f16* vT  = (f16*)take((size_t)M_ * D_ * 2);
  float* opl = (float*)take((size_t)M_ * D_ * 4);
  float* hf  = (float*)take((size_t)M_ * D_ * 4);
  f16* hc  = (f16*)take((size_t)M_ * D_ * 2);
  f16* t16 = (f16*)take((size_t)M_ * FF_ * 2);
  f16* ctx = q16;          // alias: Q dead after k_scores
  float* ffr = opl;        // alias: Oplus dead after k_ln2

  dim3 blk(256);
  k_wcvt<<<dim3((D_ * D_ + 255) / 256), blk, 0, stream>>>(Wq, wqT, D_, D_);
  k_wcvt<<<dim3((D_ * D_ + 255) / 256), blk, 0, stream>>>(Wk, wkT, D_, D_);
  k_wcvt<<<dim3((D_ * D_ + 255) / 256), blk, 0, stream>>>(Wv, wvT, D_, D_);
  k_wcvt<<<dim3((D_ * D_ + 255) / 256), blk, 0, stream>>>(Wo, woT, D_, D_);
  k_wcvt<<<dim3((D_ * FF_ + 255) / 256), blk, 0, stream>>>(W1, w1T, D_, FF_);
  k_wcvt<<<dim3((FF_ * D_ + 255) / 256), blk, 0, stream>>>(W2, w2T, FF_, D_);

  k_proj<0><<<dim3(M_ / 128, D_ / 64), blk, 0, stream>>>(x, wqT, bq, q16);
  k_proj<0><<<dim3(M_ / 128, D_ / 64), blk, 0, stream>>>(x, wkT, bk, k16);
  k_proj<1><<<dim3(M_ / 128, D_ / 64), blk, 0, stream>>>(x, wvT, bv, vT);

  k_scores<<<dim3(S_ / 64, S_ / 128, B_ * H_), blk, 0, stream>>>(q16, k16, atn);
  k_softmax<<<dim3(B_ * H_ * S_), blk, 0, stream>>>(atn);
  k_ctx<<<dim3(S_ / 128, 1, B_ * H_), blk, 0, stream>>>(atn, vT, ctx);

  k_oproj<<<dim3(M_ / 128, D_ / 64), blk, 0, stream>>>(ctx, woT, bo, x, opl);
  k_ln2<<<dim3(M_), blk, 0, stream>>>(opl, x, gmh, bmh, gn1, bn1, hf, hc);

  k_ffn1<<<dim3(M_ / 128, 1), blk, 0, stream>>>(hc, w1T, b1, t16);
  k_ffn2<<<dim3(M_ / 128, D_ / 64), blk, 0, stream>>>(t16, w2T, b2, hf, ffr);
  k_lnf<<<dim3(M_), blk, 0, stream>>>(ffr, gn2, bn2, out);

  (void)in_sizes; (void)n_in; (void)out_size; (void)ws_size;
}
